// Decoder_11355893531219
// MI455X (gfx1250) — matrix operand order repaired and checked by reading, not compiled, not run on hardware
//
#include <hip/hip_runtime.h>
#include <math.h>

// Problem constants (match reference)
#define D_    512
#define H_    8
#define DH_   64
#define DFF_  1024
#define B_    4
#define N_    1024
#define M_    (B_*N_)   // 4096 rows of activations

typedef _Float16 h8   __attribute__((ext_vector_type(8)));
typedef _Float16 v16h __attribute__((ext_vector_type(16)));
typedef float    v8f  __attribute__((ext_vector_type(8)));

union H16 { struct { h8 lo; h8 hi; } p; v16h v; };

__device__ inline v8f wmma_f16(v16h a, v16h b, v8f c) {
    // D = A(16x32 f16) * B(32x16 f16) + C(16x16 f32)
    return __builtin_amdgcn_wmma_f32_16x16x32_f16(
        /*neg_a=*/false, a, /*neg_b=*/false, b,
        /*c_mod=*/(short)0, c, /*reuse_a=*/false, /*reuse_b=*/false);
}

__device__ inline float gelu_f(float x) {
    float x3 = x*x*x;
    return 0.5f*x*(1.0f + tanhf(0.7978845608028654f*(x + 0.044715f*x3)));
}

// ---------------------------------------------------------------------------
// Tiled WMMA GEMM:  C[M,N] = act( A[M,K] @ W[K,N] + bias[N] )
// BM=128, BN=64, BK=32. 256 threads = 8 waves; wave w owns rows 16w..16w+15
// and all 64 cols (4 f32 accumulators). bias is always non-null.
// ---------------------------------------------------------------------------
__global__ __launch_bounds__(256)
void gemm_bias_kernel(const float* __restrict__ A, const float* __restrict__ W,
                      const float* __restrict__ bias, float* __restrict__ C,
                      int M, int N, int K, int act)
{
    __shared__ __align__(64) _Float16 As[128][32];   // 8 KB, row-major
    __shared__ __align__(64) _Float16 Bst[64][32];   // 4 KB, TRANSPOSED: Bst[n][k]

    const int tid  = threadIdx.x;
    const int wave = tid >> 5;
    const int lane = tid & 31;
    const int hi   = lane >> 4;   // half-wave select
    const int ln   = lane & 15;
    const int rowBase = blockIdx.y * 128;
    const int colBase = blockIdx.x * 64;

    v8f acc[4];
    #pragma unroll
    for (int t = 0; t < 4; ++t)
        #pragma unroll
        for (int r = 0; r < 8; ++r) acc[t][r] = 0.0f;

    // Per-thread staging coordinates
    const int arow = tid >> 1;            // A: 0..127
    const int acb  = (tid & 1) * 16;      // A: col base 0 or 16
    const int bkk  = tid >> 3;            // B: 0..31
    const int bnb  = (tid & 7) * 8;       // B: 0..56

    for (int kb = 0; kb < K; kb += 32) {
        // Stage A tile: 128x32 halves, 16 per thread (contiguous f32 reads)
        {
            const float* src = A + (size_t)(rowBase + arow) * K + kb + acb;
            #pragma unroll
            for (int j = 0; j < 16; ++j) As[arow][acb + j] = (_Float16)src[j];
        }
        // Stage B tile transposed: 32x64 -> Bst[n][k], 8 per thread
        {
            const float* src = W + (size_t)(kb + bkk) * N + colBase + bnb;
            #pragma unroll
            for (int j = 0; j < 8; ++j) Bst[bnb + j][bkk] = (_Float16)src[j];
        }
        // Prefetch next K-tile into cache while this tile computes
        if (kb + 32 < K) {
            __builtin_prefetch(A + (size_t)(rowBase + arow) * K + kb + 32 + acb, 0, 1);
            __builtin_prefetch(W + (size_t)(kb + 32 + bkk) * N + colBase + bnb, 0, 1);
        }
        __syncthreads();

        // A fragment: lane ln holds row m; halves = K hi*16..+15
        const int m = wave * 16 + ln;
        H16 ua;
        ua.p.lo = *(const h8*)&As[m][hi * 16];
        ua.p.hi = *(const h8*)&As[m][hi * 16 + 8];
        v16h a = ua.v;

        #pragma unroll
        for (int t = 0; t < 4; ++t) {
            // B fragment: lane ln holds col n=t*16+ln; halves = K hi*16..+15
            v16h bt = *(const v16h*)&Bst[t * 16 + ln][hi * 16];
            acc[t] = wmma_f16(a, bt, acc[t]);
        }
        __syncthreads();
    }

    // Epilogue: bias (+ GELU), store f32
    #pragma unroll
    for (int t = 0; t < 4; ++t) {
        int col = colBase + t * 16 + ln;
        float bv = bias[col];
        #pragma unroll
        for (int r = 0; r < 8; ++r) {
            float val = acc[t][r] + bv;
            if (act) val = gelu_f(val);
            C[(size_t)(rowBase + wave * 16 + r + 8 * hi) * N + col] = val;
        }
    }
}

// ---------------------------------------------------------------------------
// Flash attention with on-the-fly geometric RBF bias.
// Block = one (b, h, 128-query slice). 8 waves x 16 queries.
// Loops over keys in blocks of 32; online softmax; O accumulated via WMMA.
// ---------------------------------------------------------------------------
__global__ __launch_bounds__(256)
void flash_bias_attn(const float* __restrict__ Q, const float* __restrict__ Km,
                     const float* __restrict__ Vm, const float* __restrict__ coords,
                     const unsigned char* __restrict__ mask, float* __restrict__ O)
{
    __shared__ __align__(64) _Float16 Ks[32][64];      // 4 KB  K block [key][dh]
    __shared__ __align__(64) _Float16 Vt[64][32];      // 4 KB  V block transposed [dh][key]
    __shared__ __align__(64) _Float16 Ps[8][16][32];   // 8 KB per-wave P staging
    __shared__ float         Kc[32][3];                // key coords
    __shared__ unsigned char Kpad[32];                 // key padding mask

    const int tid  = threadIdx.x;
    const int wave = tid >> 5;
    const int lane = tid & 31;
    const int hi   = lane >> 4;
    const int ln   = lane & 15;

    const int NQB  = N_ / 128;
    const int bid  = blockIdx.x;
    const int qblk = bid % NQB;
    const int h    = (bid / NQB) % H_;
    const int b    = bid / (NQB * H_);
    const int qbase = qblk * 128 + wave * 16;

    // per-head spread: MIN_SPREAD * (MAX/MIN)^(h/7)
    const float spread = 1.0f * powf(6.0f, (float)h * (1.0f / 7.0f));
    const float inv2s2 = 1.0f / (2.0f * spread * spread);
    const float LOGMIN = -4.6051701860f;   // log(0.01)
    const float LOGMAX = -0.0100503359f;   // log(0.99)
    const float BETA   = 2.0f;

    // Q fragments (scaled by 1/sqrt(DH)=0.125), K-dim 64 split into two slices
    const float* qp = Q + ((size_t)(b * N_ + qbase + ln)) * D_ + h * DH_;
    v16h qf[2];
    #pragma unroll
    for (int s = 0; s < 2; ++s) {
        H16 u;
        #pragma unroll
        for (int j = 0; j < 8; ++j) u.p.lo[j] = (_Float16)(qp[s*32 + hi*16 + j] * 0.125f);
        #pragma unroll
        for (int j = 0; j < 8; ++j) u.p.hi[j] = (_Float16)(qp[s*32 + hi*16 + 8 + j] * 0.125f);
        qf[s] = u.v;
    }
    // q coords for the 8 rows this lane owns (rows r + 8*hi of the 16x16 tiles)
    float qc[8][3];
    #pragma unroll
    for (int r = 0; r < 8; ++r) {
        int qr = qbase + r + 8 * hi;
        qc[r][0] = coords[(size_t)(b * N_ + qr) * 3 + 0];
        qc[r][1] = coords[(size_t)(b * N_ + qr) * 3 + 1];
        qc[r][2] = coords[(size_t)(b * N_ + qr) * 3 + 2];
    }

    v8f acc[4];
    #pragma unroll
    for (int t = 0; t < 4; ++t)
        #pragma unroll
        for (int r = 0; r < 8; ++r) acc[t][r] = 0.0f;
    float rowm[8], rowl[8];
    #pragma unroll
    for (int r = 0; r < 8; ++r) { rowm[r] = -1e30f; rowl[r] = 0.0f; }

    const int key = tid >> 3;            // staging coords
    const int dh0 = (tid & 7) * 8;

    for (int kb = 0; kb < N_ / 32; ++kb) {
        // ---- stage K (row-major f16) and V (transposed f16) + coords/mask
        {
            const float* kp = Km + ((size_t)(b * N_ + kb * 32 + key)) * D_ + h * DH_ + dh0;
            const float* vp = Vm + ((size_t)(b * N_ + kb * 32 + key)) * D_ + h * DH_ + dh0;
            #pragma unroll
            for (int j = 0; j < 8; ++j) Ks[key][dh0 + j] = (_Float16)kp[j];
            #pragma unroll
            for (int j = 0; j < 8; ++j) Vt[dh0 + j][key] = (_Float16)vp[j];
            if (tid < 32) {
                Kc[tid][0] = coords[(size_t)(b * N_ + kb * 32 + tid) * 3 + 0];
                Kc[tid][1] = coords[(size_t)(b * N_ + kb * 32 + tid) * 3 + 1];
                Kc[tid][2] = coords[(size_t)(b * N_ + kb * 32 + tid) * 3 + 2];
                Kpad[tid]  = mask[b * N_ + kb * 32 + tid];
            }
            // Prefetch next key block while this one computes
            if (kb + 1 < N_ / 32) {
                __builtin_prefetch(kp + 32 * D_, 0, 1);
                __builtin_prefetch(vp + 32 * D_, 0, 1);
            }
        }
        __syncthreads();

        // ---- scores S = Q K^T / sqrt(dh) + bias, two 16x16 tiles
        float s0[8], s1[8];
        #pragma unroll
        for (int st = 0; st < 2; ++st) {
            v8f sc;
            #pragma unroll
            for (int r = 0; r < 8; ++r) sc[r] = 0.0f;
            #pragma unroll
            for (int s = 0; s < 2; ++s) {
                v16h bf = *(const v16h*)&Ks[st * 16 + ln][s * 32 + hi * 16];
                sc = wmma_f16(qf[s], bf, sc);
            }
            int   kl = st * 16 + ln;
            float kx = Kc[kl][0], ky = Kc[kl][1], kz = Kc[kl][2];
            float mterm = Kpad[kl] ? -1e9f : 0.0f;
            float* sd = st ? s1 : s0;
            #pragma unroll
            for (int r = 0; r < 8; ++r) {
                float dx = qc[r][0] - kx, dy = qc[r][1] - ky, dz = qc[r][2] - kz;
                float dd = dx*dx + dy*dy + dz*dz;
                float lb = -dd * inv2s2;
                lb = fminf(fmaxf(lb, LOGMIN), LOGMAX) * BETA;
                sd[r] = sc[r] + lb + mterm;
            }
        }

        // ---- online softmax update (row reductions within 16-lane groups)
        #pragma unroll
        for (int r = 0; r < 8; ++r) {
            float mx = fmaxf(s0[r], s1[r]);
            mx = fmaxf(mx, __shfl_xor(mx, 1, 16));
            mx = fmaxf(mx, __shfl_xor(mx, 2, 16));
            mx = fmaxf(mx, __shfl_xor(mx, 4, 16));
            mx = fmaxf(mx, __shfl_xor(mx, 8, 16));
            float newm = fmaxf(rowm[r], mx);
            float scl  = __expf(rowm[r] - newm);
            float p0   = __expf(s0[r] - newm);
            float p1   = __expf(s1[r] - newm);
            float rs   = p0 + p1;
            rs += __shfl_xor(rs, 1, 16);
            rs += __shfl_xor(rs, 2, 16);
            rs += __shfl_xor(rs, 4, 16);
            rs += __shfl_xor(rs, 8, 16);
            rowl[r] = rowl[r] * scl + rs;
            rowm[r] = newm;
            #pragma unroll
            for (int t = 0; t < 4; ++t) acc[t][r] *= scl;
            // stage P (C-layout scatter -> per-wave LDS tile)
            Ps[wave][r + 8 * hi][ln]      = (_Float16)p0;
            Ps[wave][r + 8 * hi][16 + ln] = (_Float16)p1;
        }

        // ---- O += P @ V (reload P in A-layout; same-wave LDS ordering)
        H16 up;
        up.p.lo = *(const h8*)&Ps[wave][ln][hi * 16];
        up.p.hi = *(const h8*)&Ps[wave][ln][hi * 16 + 8];
        v16h pf = up.v;
        #pragma unroll
        for (int t = 0; t < 4; ++t) {
            v16h vf = *(const v16h*)&Vt[t * 16 + ln][hi * 16];
            acc[t] = wmma_f16(pf, vf, acc[t]);
        }
        __syncthreads();
    }

    // ---- normalize and store (heads concatenated: [B,N,H*DH])
    #pragma unroll
    for (int r = 0; r < 8; ++r) {
        float inv = 1.0f / rowl[r];
        #pragma unroll
        for (int t = 0; t < 4; ++t) {
            O[((size_t)(b * N_ + qbase + r + 8 * hi)) * D_ + h * DH_ + t * 16 + ln]
                = acc[t][r] * inv;
        }
    }
}

// ---------------------------------------------------------------------------
// Fused residual add + LayerNorm over rows of length D_=512.
// Block = one row, 256 threads x 2 elements. Safe for y aliasing x or dlt.
// ---------------------------------------------------------------------------
__global__ __launch_bounds__(256)
void add_ln_kernel(const float* __restrict__ x, const float* __restrict__ dlt,
                   const float* __restrict__ g, const float* __restrict__ bta,
                   float* __restrict__ y)
{
    const int row = blockIdx.x;
    const int tid = threadIdx.x;
    const int wave = tid >> 5, lane = tid & 31;
    const float* xr = x + (size_t)row * D_;
    const float* dr = dlt + (size_t)row * D_;
    float v0 = xr[tid] + dr[tid];
    float v1 = xr[tid + 256] + dr[tid + 256];

    __shared__ float part[8];
    float s = v0 + v1;
    for (int m = 16; m >= 1; m >>= 1) s += __shfl_xor(s, m, 32);
    if (lane == 0) part[wave] = s;
    __syncthreads();
    float tot = 0.0f;
    #pragma unroll
    for (int w2 = 0; w2 < 8; ++w2) tot += part[w2];
    float mean = tot * (1.0f / D_);

    float c0 = v0 - mean, c1 = v1 - mean;
    float qv = c0 * c0 + c1 * c1;
    for (int m = 16; m >= 1; m >>= 1) qv += __shfl_xor(qv, m, 32);
    __syncthreads();
    if (lane == 0) part[wave] = qv;
    __syncthreads();
    float totq = 0.0f;
    #pragma unroll
    for (int w2 = 0; w2 < 8; ++w2) totq += part[w2];
    float inv = rsqrtf(totq * (1.0f / D_) + 1e-5f);

    y[(size_t)row * D_ + tid]       = c0 * inv * g[tid]       + bta[tid];
    y[(size_t)row * D_ + tid + 256] = c1 * inv * g[tid + 256] + bta[tid + 256];
}

// ---------------------------------------------------------------------------
extern "C" void kernel_launch(void* const* d_in, const int* in_sizes, int n_in,
                              void* d_out, int out_size, void* d_ws, size_t ws_size,
                              hipStream_t stream)
{
    const float* wf0    = (const float*)d_in[0];
    const float* aas0   = (const float*)d_in[1];
    const float* coords = (const float*)d_in[2];
    const unsigned char* mask = (const unsigned char*)d_in[3];

    // Flattened params (dict insertion order):
    // per attn: wq wk wv wo bq bk bv bo | ln: g b | ffn: w1 b1 w2 b2 | ln: g b
    const float* P[32];
    for (int i = 0; i < 32 && (4 + i) < n_in; ++i) P[i] = (const float*)d_in[4 + i];
    enum { WQ=0, WK, WV, WO, BQ, BK, BV, BO, L1G, L1B, W1, B1, W2, B2, L2G, L2B, STR=16 };

    // Workspace layout (floats)
    float* q    = (float*)d_ws;
    float* k    = q    + (size_t)M_ * D_;
    float* v    = k    + (size_t)M_ * D_;
    float* att  = v    + (size_t)M_ * D_;
    float* proj = att  + (size_t)M_ * D_;
    float* hbuf = proj + (size_t)M_ * D_;     // M_ x DFF_

    float* out_wf  = (float*)d_out;
    float* out_aas = out_wf + (size_t)M_ * D_;

    auto gemm = [&](const float* A, const float* W, const float* bias, float* C,
                    int Nn, int K, int act) {
        dim3 grid(Nn / 64, M_ / 128);
        gemm_bias_kernel<<<grid, 256, 0, stream>>>(A, W, bias, C, M_, Nn, K, act);
    };
    const dim3 attn_grid(B_ * H_ * (N_ / 128));

    // ===== Layer 1: seq queries struct (Q=aas, K=wf, V=wf) =====
    gemm(aas0, P[WQ], P[BQ], q, D_, D_, 0);
    gemm(wf0,  P[WK], P[BK], k, D_, D_, 0);
    gemm(wf0,  P[WV], P[BV], v, D_, D_, 0);
    flash_bias_attn<<<attn_grid, 256, 0, stream>>>(q, k, v, coords, mask, att);
    gemm(att, P[WO], P[BO], proj, D_, D_, 0);
    add_ln_kernel<<<M_, 256, 0, stream>>>(aas0, proj, P[L1G], P[L1B], out_aas);
    gemm(out_aas, P[W1], P[B1], hbuf, DFF_, D_, 1);          // GELU
    gemm(hbuf,    P[W2], P[B2], proj, D_, DFF_, 0);
    add_ln_kernel<<<M_, 256, 0, stream>>>(out_aas, proj, P[L2G], P[L2B], out_aas);

    // ===== Layer 2: struct queries seq (Q=wf, K=aas_new, V=wf) =====
    gemm(wf0,     P[STR+WQ], P[STR+BQ], q, D_, D_, 0);
    gemm(out_aas, P[STR+WK], P[STR+BK], k, D_, D_, 0);
    gemm(wf0,     P[STR+WV], P[STR+BV], v, D_, D_, 0);
    flash_bias_attn<<<attn_grid, 256, 0, stream>>>(q, k, v, coords, mask, att);
    gemm(att, P[STR+WO], P[STR+BO], proj, D_, D_, 0);
    add_ln_kernel<<<M_, 256, 0, stream>>>(wf0, proj, P[STR+L1G], P[STR+L1B], out_wf);
    gemm(out_wf, P[STR+W1], P[STR+B1], hbuf, DFF_, D_, 1);   // GELU
    gemm(hbuf,   P[STR+W2], P[STR+B2], proj, D_, DFF_, 0);
    add_ln_kernel<<<M_, 256, 0, stream>>>(out_wf, proj, P[STR+L2G], P[STR+L2B], out_wf);

    (void)in_sizes; (void)out_size; (void)ws_size;
}
